// GATESynergy_89455578841355
// MI455X (gfx1250) — matrix operand order repaired and checked by reading, not compiled, not run on hardware
//
#include <hip/hip_runtime.h>

#define DIN     128
#define DMODEL  128
#define NMAX    160
#define NHEAD   4
#define DH      32
#define NGRAPH  512
#define NWAVE   8
#define BLOCK   256

typedef __attribute__((ext_vector_type(16))) __bf16 v16bf;
typedef __attribute__((ext_vector_type(8)))  __bf16 v8bf;
typedef __attribute__((ext_vector_type(8)))  float  v8f;
typedef __attribute__((ext_vector_type(4)))  float  v4f;

__device__ __forceinline__ v16bf join8(v8bf lo, v8bf hi) {
  return __builtin_shufflevector(lo, hi, 0,1,2,3,4,5,6,7,8,9,10,11,12,13,14,15);
}

__device__ __forceinline__ v8f wmma_bf16(v16bf a, v16bf b, v8f c) {
  return __builtin_amdgcn_wmma_f32_16x16x32_bf16(false, a, false, b, (short)0, c,
                                                 false, false);
}

// A-matrix fragment (16x32 MxK) from row-major bf16 LDS, row stride ld (halves).
// Lane layout: M = lane%16; halves 0..7 -> K = 8*khi + t; halves 8..15 -> K = 16 + 8*khi + t.
__device__ __forceinline__ v16bf load_A_lds(const __bf16* base, int ld, int lane) {
  const int m = lane & 15, khi = (lane >> 4) & 1;
  const __bf16* rp = base + m * ld + khi * 8;
  v8bf lo = *(const v8bf*)rp;          // K = 8*khi + 0..7
  v8bf hi = *(const v8bf*)(rp + 16);   // K = 16 + 8*khi + 0..7
  return join8(lo, hi);
}

// A fragment from global f32 x (row-major, 128 cols), rows clamped to [0, nrows)
__device__ __forceinline__ v16bf load_A_xg(const float* x, long row0, int nrows,
                                           int k0, int lane) {
  const int m = lane & 15, khi = (lane >> 4) & 1;
  long row = row0 + m;
  if (row >= nrows) row = nrows - 1;
  const v4f* rp4 = (const v4f*)(x + row * DIN + k0 + khi * 8);
  v4f a0 = rp4[0], a1 = rp4[1];   // K = k0 + 8*khi + 0..7
  v4f a2 = rp4[4], a3 = rp4[5];   // K = k0 + 16 + 8*khi + 0..7
  v16bf a;
  #pragma unroll
  for (int t = 0; t < 4; ++t) {
    a[t]      = (__bf16)a0[t];
    a[4 + t]  = (__bf16)a1[t];
    a[8 + t]  = (__bf16)a2[t];
    a[12 + t] = (__bf16)a3[t];
  }
  return a;
}

// B fragment (32x16 KxN): B[k][n] = W[n0+n][k0+k], W is f32 row-major [128][128].
// Lane layout: N = lane%16; halves 0..7 -> K = kg + t; halves 8..15 -> K = 16 + kg + t.
__device__ __forceinline__ v16bf load_B_w(const float* W, int n0, int k0, int lane) {
  const int n = lane & 15, kg = ((lane >> 4) & 1) * 8;
  const v4f* rp4 = (const v4f*)(W + (long)(n0 + n) * DIN + k0 + kg);
  v4f q0 = rp4[0], q1 = rp4[1], q2 = rp4[4], q3 = rp4[5];
  v16bf b;
  #pragma unroll
  for (int t = 0; t < 4; ++t) {
    b[t]      = (__bf16)q0[t];
    b[4 + t]  = (__bf16)q1[t];
    b[8 + t]  = (__bf16)q2[t];
    b[12 + t] = (__bf16)q3[t];
  }
  return b;
}

// B fragment where B[k][n] = M[n0+n][k0+k], M bf16 LDS row-major (contiguous in k).
__device__ __forceinline__ v16bf load_B_t(const __bf16* M, int ld, int n0, int k0,
                                          int lane) {
  const int n = lane & 15, kg = ((lane >> 4) & 1) * 8;
  const __bf16* rp = M + (n0 + n) * ld + k0 + kg;
  v8bf lo = *(const v8bf*)rp;
  v8bf hi = *(const v8bf*)(rp + 16);
  return join8(lo, hi);
}

__global__ __launch_bounds__(BLOCK)
void ga_attn_kernel(const float* __restrict__ x,  const int* __restrict__ batch,
                    const float* __restrict__ Wq, const float* __restrict__ bq,
                    const float* __restrict__ Wk, const float* __restrict__ bk,
                    const float* __restrict__ Wv, const float* __restrict__ bv,
                    const float* __restrict__ Wo, const float* __restrict__ bo,
                    float* __restrict__ out, int T) {
  // LDS: 4 * 40960 = 163840 bytes exactly (2 workgroups fit in a 320KB WGP)
  __shared__ __align__(16) __bf16 Qs[NMAX * DMODEL];   // Q row-major [row][d]
  __shared__ __align__(16) __bf16 Ks[NMAX * DMODEL];   // K row-major [row][d]
  __shared__ __align__(16) __bf16 Vt[DMODEL * NMAX];   // V transposed [d][row]
  __shared__ __align__(16) __bf16 Pb[NWAVE * 16 * NMAX];

  const int tid  = threadIdx.x;
  const int wave = tid >> 5;
  const int lane = tid & 31;
  const int g    = blockIdx.x;

  // graph start/count via binary search on sorted `batch`; stash in head of Pb
  int* sc = (int*)Pb;
  if (tid == 0) {
    int lo = 0, hi = T;
    while (lo < hi) { int mid = (lo + hi) >> 1; if (batch[mid] <  g) lo = mid + 1; else hi = mid; }
    const int s0 = lo;
    hi = T;
    while (lo < hi) { int mid = (lo + hi) >> 1; if (batch[mid] <= g) lo = mid + 1; else hi = mid; }
    sc[0] = s0;
    sc[1] = lo - s0;
  }
  __syncthreads();
  const int start = sc[0];
  const int n     = sc[1];
  const int nrb   = (n + 15) >> 4;  // 16-wide key/query tiles
  const int nkt   = (n + 31) >> 5;  // 32-deep K steps for P@V
  const int nrbp  = nkt * 2;        // row blocks written in phase 1 (pads K/V for PV reads)

  const int nn  = lane & 15;
  const int khi = (lane >> 4) & 1;

  // ---------------- Phase 1: Q/K/V = X*W^T + b  ->  bf16 LDS ----------------
  for (int rb = wave; rb < nrbp; rb += NWAVE) {
    v16bf afr[4];
    #pragma unroll
    for (int ks = 0; ks < 4; ++ks)
      afr[ks] = load_A_xg(x, (long)start + rb * 16, T, ks * 32, lane);
    #pragma unroll
    for (int mtx = 0; mtx < 3; ++mtx) {
      const float* W  = (mtx == 0) ? Wq : (mtx == 1) ? Wk : Wv;
      const float* bb = (mtx == 0) ? bq : (mtx == 1) ? bk : bv;
      for (int ct = 0; ct < 8; ++ct) {
        const int col0 = ct * 16;
        v8f acc = {};
        #pragma unroll
        for (int ks = 0; ks < 4; ++ks)
          acc = wmma_bf16(afr[ks], load_B_w(W, col0, ks * 32, lane), acc);
        const float bias = bb[col0 + nn];
        if (mtx == 2) {
          // V: store transposed; rows of the C tile are consecutive in Vt -> 1x b128
          v8bf pk;
          #pragma unroll
          for (int r = 0; r < 8; ++r) pk[r] = (__bf16)(acc[r] + bias);
          *(v8bf*)(Vt + (col0 + nn) * NMAX + rb * 16 + 8 * khi) = pk;
        } else {
          __bf16* dst = (mtx == 0) ? Qs : Ks;
          #pragma unroll
          for (int r = 0; r < 8; ++r)
            dst[(rb * 16 + r + 8 * khi) * DMODEL + col0 + nn] = (__bf16)(acc[r] + bias);
        }
      }
    }
  }
  __syncthreads();

  // ------- Phase 2: fused attention + output projection per query tile ------
  const float scale = 0.17677669529663687f;  // 1/sqrt(DH)
  __bf16* Pw = Pb + wave * (16 * NMAX);      // per-wave staging buffer

  for (int qb = wave; qb < nrb; qb += NWAVE) {
    // zero P buffer once per task (keeps masked key columns at exactly 0)
    {
      const v8bf z8 = {};
      #pragma unroll
      for (int i = 0; i < 10; ++i)
        *(v8bf*)(Pw + (lane + 32 * i) * 8) = z8;   // 2560 halves, packed b128 stores
    }

    const v8f vz = {};
    v8f o[8];                                // O tiles: [head][2 x 16-col tiles]
    #pragma unroll
    for (int i = 0; i < 8; ++i) o[i] = vz;

    #pragma unroll
    for (int h = 0; h < NHEAD; ++h) {
      const int hb = h * DH;
      const v16bf qf = load_A_lds(Qs + qb * 16 * DMODEL + hb, DMODEL, lane);

      v8f s[10];                             // full score row block (16 x up-to-160)
      #pragma unroll
      for (int j = 0; j < 10; ++j)
        if (j < nrb)
          s[j] = wmma_bf16(qf, load_B_t(Ks, DMODEL, j * 16, hb, lane), vz);

      // masked softmax across keys (row = query; reduce across 16-lane half)
      float mx[8], sm[8];
      #pragma unroll
      for (int r = 0; r < 8; ++r) { mx[r] = -3.0e38f; sm[r] = 0.0f; }
      #pragma unroll
      for (int j = 0; j < 10; ++j)
        if (j < nrb) {
          const bool valid = (j * 16 + nn) < n;
          #pragma unroll
          for (int r = 0; r < 8; ++r)
            if (valid) mx[r] = fmaxf(mx[r], s[j][r] * scale);
        }
      #pragma unroll
      for (int r = 0; r < 8; ++r) {
        float m = mx[r];
        m = fmaxf(m, __shfl_xor(m, 1, 32));
        m = fmaxf(m, __shfl_xor(m, 2, 32));
        m = fmaxf(m, __shfl_xor(m, 4, 32));
        m = fmaxf(m, __shfl_xor(m, 8, 32));
        mx[r] = m;
      }
      #pragma unroll
      for (int j = 0; j < 10; ++j)
        if (j < nrb) {
          const bool valid = (j * 16 + nn) < n;
          #pragma unroll
          for (int r = 0; r < 8; ++r) {
            float e = valid ? __expf(s[j][r] * scale - mx[r]) : 0.0f;
            s[j][r] = e;
            sm[r] += e;
          }
        }
      #pragma unroll
      for (int r = 0; r < 8; ++r) {
        float t = sm[r];
        t += __shfl_xor(t, 1, 32);
        t += __shfl_xor(t, 2, 32);
        t += __shfl_xor(t, 4, 32);
        t += __shfl_xor(t, 8, 32);
        sm[r] = 1.0f / t;
      }
      // store P (bf16) to per-wave buffer, row-major [16][NMAX]
      #pragma unroll
      for (int j = 0; j < 10; ++j)
        if (j < nrb) {
          #pragma unroll
          for (int r = 0; r < 8; ++r)
            Pw[(r + 8 * khi) * NMAX + j * 16 + nn] = (__bf16)(s[j][r] * sm[r]);
        }
      // O_h += P @ V_h  (K in steps of 32; P is 0 beyond n; Vt gives contiguous B frags)
      for (int kt = 0; kt < nkt; ++kt) {
        const v16bf pf = load_A_lds(Pw + kt * 32, NMAX, lane);
        o[2 * h + 0] = wmma_bf16(pf, load_B_t(Vt, NMAX, hb,      kt * 32, lane), o[2 * h + 0]);
        o[2 * h + 1] = wmma_bf16(pf, load_B_t(Vt, NMAX, hb + 16, kt * 32, lane), o[2 * h + 1]);
      }
    }

    // stage O rows (16 x 128 bf16) into the per-wave buffer for the Wo matmul
    #pragma unroll
    for (int h = 0; h < NHEAD; ++h)
      #pragma unroll
      for (int c = 0; c < 2; ++c)
        #pragma unroll
        for (int r = 0; r < 8; ++r)
          Pw[(r + 8 * khi) * DMODEL + h * DH + c * 16 + nn] = (__bf16)o[2 * h + c][r];

    v16bf afr[4];
    #pragma unroll
    for (int ks = 0; ks < 4; ++ks)
      afr[ks] = load_A_lds(Pw + ks * 32, DMODEL, lane);

    for (int ct = 0; ct < 8; ++ct) {
      const int col0 = ct * 16;
      v8f acc = {};
      #pragma unroll
      for (int ks = 0; ks < 4; ++ks)
        acc = wmma_bf16(afr[ks], load_B_w(Wo, col0, ks * 32, lane), acc);
      const float bias = bo[col0 + nn];
      #pragma unroll
      for (int r = 0; r < 8; ++r) {
        const int row = qb * 16 + r + 8 * khi;
        if (row < n) {
          const long gi = (long)(start + row) * DMODEL + col0 + nn;
          out[gi] = acc[r] + bias + x[gi];   // + residual
        }
      }
    }
  }
}

extern "C" void kernel_launch(void* const* d_in, const int* in_sizes, int n_in,
                              void* d_out, int out_size, void* d_ws, size_t ws_size,
                              hipStream_t stream) {
  const float* x     = (const float*)d_in[0];
  const int*   batch = (const int*)  d_in[1];
  const float* Wq    = (const float*)d_in[2];
  const float* bq    = (const float*)d_in[3];
  const float* Wk    = (const float*)d_in[4];
  const float* bk    = (const float*)d_in[5];
  const float* Wv    = (const float*)d_in[6];
  const float* bv    = (const float*)d_in[7];
  const float* Wo    = (const float*)d_in[8];
  const float* bo    = (const float*)d_in[9];
  float* out = (float*)d_out;
  const int T = in_sizes[0] / DIN;

  ga_attn_kernel<<<NGRAPH, BLOCK, 0, stream>>>(x, batch, Wq, bq, Wk, bk,
                                               Wv, bv, Wo, bo, out, T);
}
